// FCOSPostProcessor_52269751993061
// MI455X (gfx1250) — 8-bit matrix operand order repaired and checked by reading, compiled, not run on hardware
//
#include <hip/hip_runtime.h>
#include <math.h>

#define NCLS   80
#define KTOP   1000
#define KPAD   1024
#define NBUK   8192
#define POSTN  100
#define NMS_T  0.6f
#define PRE_T  0.05f
#define NTILE  63

typedef __attribute__((ext_vector_type(8))) int v8i;
typedef int v4si __attribute__((vector_size(16)));
typedef __attribute__((address_space(1))) v4si* gv4p;
typedef __attribute__((address_space(3))) v4si* lv4p;

__device__ __forceinline__ float sigm(float x) { return 1.0f / (1.0f + __expf(-x)); }

__global__ void k_init(int* hist, int* params, float* cand_s, int* cand_i, int B) {
    int i = blockIdx.x * blockDim.x + threadIdx.x;
    if (i < B * NBUK) hist[i] = 0;
    if (i < B * 8)    params[i] = 0;
    if (i < B * KPAD) { cand_s[i] = -1e30f; cand_i[i] = 0; }
}

__global__ void k_hist(const float* __restrict__ cls, const float* __restrict__ ctr,
                       int* __restrict__ hist, int L) {
    __shared__ int lh[NBUK];
    int b = blockIdx.y;
    int tid = threadIdx.x;
    for (int i = tid; i < NBUK; i += blockDim.x) lh[i] = 0;
    __syncthreads();
    int loc = blockIdx.x * blockDim.x + tid;
    if (loc < L) {
        float cs = sigm(ctr[(size_t)b * L + loc]);
        const float* cp = cls + ((size_t)b * L + loc) * NCLS;
        for (int c = 0; c < NCLS; c++) {
            float s = sigm(cp[c]);
            if (s > PRE_T) {
                float v = s * cs;
                unsigned bk = __float_as_uint(v) >> 17;
                if (bk >= NBUK) bk = NBUK - 1;
                atomicAdd(&lh[bk], 1);
            }
        }
    }
    __syncthreads();
    for (int i = tid; i < NBUK; i += blockDim.x)
        if (lh[i]) atomicAdd(&hist[b * NBUK + i], lh[i]);
}

__global__ void k_scan(const int* __restrict__ hist, int* __restrict__ params, int B) {
    int b = threadIdx.x;
    if (b >= B) return;
    const int* h = hist + b * NBUK;
    int cum = 0, T = -1, above = 0;
    for (int bk = NBUK - 1; bk >= 0; --bk) {
        int c = h[bk];
        if (cum + c >= KTOP) { T = bk; above = cum; break; }
        cum += c;
    }
    int remaining;
    if (T < 0) { above = cum; remaining = 0; }
    else       remaining = KTOP - above;
    int* p = params + b * 8;
    p[0] = T; p[1] = above; p[2] = remaining;
}

__global__ void k_compact(const float* __restrict__ cls, const float* __restrict__ ctr,
                          int* params, float* cand_s, int* cand_i, int L) {
    int b = blockIdx.y;
    int loc = blockIdx.x * blockDim.x + threadIdx.x;
    if (loc >= L) return;
    int T = params[b * 8 + 0];
    int above = params[b * 8 + 1];
    int remaining = params[b * 8 + 2];
    float cs = sigm(ctr[(size_t)b * L + loc]);
    const float* cp = cls + ((size_t)b * L + loc) * NCLS;
    for (int c = 0; c < NCLS; c++) {
        float s = sigm(cp[c]);
        if (s <= PRE_T) continue;
        float v = s * cs;
        int bk = (int)(__float_as_uint(v) >> 17);
        if (bk >= NBUK) bk = NBUK - 1;
        int slot = -1;
        if (bk > T) {
            slot = atomicAdd(&params[b * 8 + 3], 1);
        } else if (bk == T) {
            int e = atomicAdd(&params[b * 8 + 4], 1);
            if (e < remaining) slot = above + e;
        }
        if (slot >= 0 && slot < KTOP) {
            cand_s[b * KPAD + slot] = v;
            cand_i[b * KPAD + slot] = loc * NCLS + c;
        }
    }
}

__global__ void k_sort(float* cand_s, int* cand_i) {
    __shared__ float ss[KPAD];
    __shared__ int   ii[KPAD];
    int b = blockIdx.x, tid = threadIdx.x;
    for (int i = tid; i < KPAD; i += blockDim.x) { ss[i] = cand_s[b * KPAD + i]; ii[i] = cand_i[b * KPAD + i]; }
    for (unsigned k = 2; k <= KPAD; k <<= 1)
        for (unsigned j = k >> 1; j > 0; j >>= 1) {
            __syncthreads();
            for (unsigned t = tid; t < KPAD; t += blockDim.x) {
                unsigned x = t ^ j;
                if (x > t) {
                    float s1 = ss[t], s2 = ss[x];
                    int a1 = ii[t], a2 = ii[x];
                    bool firstTop = (s1 > s2) || (s1 == s2 && a1 < a2);
                    bool desc = ((t & k) == 0);
                    if (desc ? !firstTop : firstTop) { ss[t] = s2; ss[x] = s1; ii[t] = a2; ii[x] = a1; }
                }
            }
        }
    __syncthreads();
    for (int i = tid; i < KPAD; i += blockDim.x) { cand_s[b * KPAD + i] = ss[i]; cand_i[b * KPAD + i] = ii[i]; }
}

__global__ void __launch_bounds__(256) k_nms(
    const float* __restrict__ locs, const float* __restrict__ reg,
    const float* __restrict__ cand_s, const int* __restrict__ cand_i,
    float* __restrict__ out, int L, int B)
{
    __shared__ float4   sbox[KPAD];
    __shared__ int      scls[KPAD];
    __shared__ float    sscore[KPAD];
    __shared__ unsigned smat[KTOP * 32];
    __shared__ unsigned svalid[32];
    __shared__ unsigned skeepw[32];
    __shared__ int      skeep[POSTN];
    __shared__ int      snkeep;

    int b = blockIdx.x;
    int tid = threadIdx.x;
    int lane = tid & 31;

    float* stage_s = (float*)smat;
    int*   stage_i = (int*)(smat + KPAD);
#if defined(__AMDGCN__) && __has_builtin(__builtin_amdgcn_global_load_async_to_lds_b128)
    {
        const char* gs = (const char*)(cand_s + (size_t)b * KPAD);
        const char* gi = (const char*)(cand_i + (size_t)b * KPAD);
        __builtin_amdgcn_global_load_async_to_lds_b128(
            (gv4p)(gs + (size_t)tid * 16),
            (lv4p)((char*)stage_s + (size_t)tid * 16), 0, 0);
        __builtin_amdgcn_global_load_async_to_lds_b128(
            (gv4p)(gi + (size_t)tid * 16),
            (lv4p)((char*)stage_i + (size_t)tid * 16), 0, 0);
#if __has_builtin(__builtin_amdgcn_s_wait_asynccnt)
        __builtin_amdgcn_s_wait_asynccnt(0);
#endif
    }
#else
    for (int i = tid; i < KPAD; i += 256) { stage_s[i] = cand_s[b * KPAD + i]; stage_i[i] = cand_i[b * KPAD + i]; }
#endif
    __syncthreads();

    for (int i = tid; i < KPAD; i += 256) {
        float s = stage_s[i];
        int idx = stage_i[i];
        int loc = idx / NCLS;
        int c   = idx - loc * NCLS;
        float px = locs[loc * 2 + 0], py = locs[loc * 2 + 1];
        const float* rp = reg + ((size_t)b * L + loc) * 4;
        sbox[i] = make_float4(px - rp[0], py - rp[1], px + rp[2], py + rp[3]);
        scls[i] = c;
        sscore[i] = s;
    }
    __syncthreads();

    if (tid < 32) {
        unsigned w = 0;
        for (int q = 0; q < 32; q++) {
            int i = tid * 32 + q;
            if (i < KTOP && sscore[i] > 0.0f) w |= 1u << q;
        }
        svalid[tid] = w;
    }
    for (int i = tid; i < KTOP * 32; i += 256) smat[i] = 0;
    __syncthreads();

    int wave = tid >> 5;
    int halfsel = (lane >= 16) ? 1 : 0;
    int lm = lane & 15;
    for (int t = wave; t < NTILE * NTILE; t += 8) {
        int ti = t / NTILE, tj = t % NTILE;
        if (tj < ti) continue;
        int rowc = scls[ti * 16 + lm];
        int colc = scls[tj * 16 + lm];
        v8i acc = {0, 0, 0, 0, 0, 0, 0, 0};
        for (int ch = 0; ch < 2; ++ch) {
            int base = ch * 64;
            int kk  = rowc - base;
            int inA = (kk >= 0) & (kk < 64);
            int rem = kk & 15;
            int va  = ((kk >> 4) & 3) * 2 + ((rem >> 2) & 1);
            unsigned bva = (inA && ((rem >> 3) == halfsel)) ? (1u << ((rem & 3) * 8)) : 0u;
            v8i A;
            for (int q = 0; q < 8; q++) A[q] = (q == va) ? (int)bva : 0;
            int kb  = colc - base;
            int inB = (kb >= 0) & (kb < 64);
            int r32 = kb & 15;
            int vb  = ((kb >> 4) & 3) * 2 + ((r32 >> 2) & 1);
            unsigned bvb = (inB && ((r32 >> 3) == halfsel)) ? (1u << ((r32 & 3) * 8)) : 0u;
            v8i Bm;
            for (int q = 0; q < 8; q++) Bm[q] = (q == vb) ? (int)bvb : 0;
            acc = __builtin_amdgcn_wmma_i32_16x16x64_iu8(false, A, false, Bm, acc, false, false);
        }
        int ncol = tj * 16 + lm;
        for (int r = 0; r < 8; r++) {
            int row = ti * 16 + r + halfsel * 8;
            if (acc[r] != 0 && row < KTOP && ncol < KTOP && ncol > row) {
                float4 a = sbox[row], c = sbox[ncol];
                float areaA = (a.z - a.x) * (a.w - a.y);
                float areaC = (c.z - c.x) * (c.w - c.y);
                float iw = fminf(a.z, c.z) - fmaxf(a.x, c.x);
                float ih = fminf(a.w, c.w) - fmaxf(a.y, c.y);
                float inter = fmaxf(iw, 0.0f) * fmaxf(ih, 0.0f);
                float iou = inter / (areaA + areaC - inter + 1e-9f);
                if (iou > NMS_T) atomicOr(&smat[row * 32 + (ncol >> 5)], 1u << (ncol & 31));
            }
        }
    }
    __syncthreads();

    if (tid < 32) {
        unsigned supp = 0;
        unsigned vw = svalid[lane];
        for (int i = 0; i < KTOP; i++) {
            unsigned ws = __shfl(supp, i >> 5, 32);
            unsigned wv = __shfl(vw,   i >> 5, 32);
            int bit = i & 31;
            bool alive = (((ws >> bit) & 1u) == 0u) && (((wv >> bit) & 1u) != 0u);
            if (alive) supp |= smat[i * 32 + lane];
        }
        skeepw[lane] = vw & ~supp;
    }
    __syncthreads();
    if (tid == 0) {
        int n = 0;
        for (int i = 0; i < KTOP && n < POSTN; i++)
            if ((skeepw[i >> 5] >> (i & 31)) & 1u) skeep[n++] = i;
        snkeep = n;
    }
    __syncthreads();

    float* ob = out;
    float* os = out + (size_t)B * POSTN * 4;
    float* oc = os + (size_t)B * POSTN;
    float* ov = oc + (size_t)B * POSTN;
    for (int o = tid; o < POSTN; o += 256) {
        float4 bx = make_float4(0.f, 0.f, 0.f, 0.f);
        float sc = 0.f, cl = 0.f, vv = 0.f;
        if (o < snkeep) {
            int i = skeep[o];
            bx = sbox[i];
            sc = sqrtf(fmaxf(sscore[i], 0.f));
            cl = (float)scls[i];
            vv = 1.0f;
        }
        float* bp = ob + ((size_t)b * POSTN + o) * 4;
        bp[0] = bx.x; bp[1] = bx.y; bp[2] = bx.z; bp[3] = bx.w;
        os[b * POSTN + o] = sc;
        oc[b * POSTN + o] = cl;
        ov[b * POSTN + o] = vv;
    }
}

extern "C" void kernel_launch(void* const* d_in, const int* in_sizes, int n_in,
                              void* d_out, int out_size, void* d_ws, size_t ws_size,
                              hipStream_t stream) {
    (void)n_in; (void)out_size; (void)ws_size;
    const float* locs = (const float*)d_in[0];
    const float* cls  = (const float*)d_in[1];
    const float* reg  = (const float*)d_in[2];
    const float* ctr  = (const float*)d_in[3];
    int L = in_sizes[0] / 2;
    int B = in_sizes[3] / L;

    char* ws = (char*)d_ws;
    size_t off = 0;
    int* hist = (int*)(ws + off);      off += (size_t)B * NBUK * 4;
    int* params = (int*)(ws + off);    off += (size_t)B * 8 * 4;   off = (off + 255) & ~(size_t)255;
    float* cand_s = (float*)(ws + off); off += (size_t)B * KPAD * 4;
    int* cand_i = (int*)(ws + off);
    float* out = (float*)d_out;

    int initN = B * NBUK;
    k_init<<<(initN + 255) / 256, 256, 0, stream>>>(hist, params, cand_s, cand_i, B);
    dim3 g((L + 255) / 256, B);
    k_hist<<<g, 256, 0, stream>>>(cls, ctr, hist, L);
    k_scan<<<1, 32, 0, stream>>>(hist, params, B);
    k_compact<<<g, 256, 0, stream>>>(cls, ctr, params, cand_s, cand_i, L);
    k_sort<<<B, 256, 0, stream>>>(cand_s, cand_i);
    k_nms<<<B, 256, 0, stream>>>(locs, reg, cand_s, cand_i, out, L, B);
}
